// MPNN_79010218377377
// MI455X (gfx1250) — compile-verified, matrix operand order checked by reading, not run on hardware
//
#include <hip/hip_runtime.h>

// Problem constants (match reference)
#define NN      500000
#define EE      4000000
#define GG      25000
#define ATOMD   64
#define AH      16
#define BH      8
#define NODE_H  80
#define EDGE_H  24
#define OUTD    2
#define NLAYER  4

typedef __attribute__((ext_vector_type(16))) _Float16 v16h;
typedef __attribute__((ext_vector_type(8)))  _Float16 v8h;
typedef __attribute__((ext_vector_type(8)))  float    v8f;

// Pre-swizzled B-operand table layout (tile index -> 32 lanes of v16h)
#define BT_NW1   0   // 10: [kc(2)][nt(5)]
#define BT_NW2  10   // 3:  kc 0..2
#define BT_EW1  13   // 2:  nt 0..1
#define BT_EW2  15   // 1
#define BT_LIN  16   // 1
#define BT_GW1  17   // 2
#define BT_GW2  19   // 1
#define BT_WIH  20   // 3
#define BT_WHH  23   // 3
#define BT_UW1  26   // 2
#define BT_UW2  28   // 2
#define BT_RW1  30   // 4
#define BT_RW2  34   // 8: [kc(2)][nt(4)]
#define BT_CW1  42   // 4
#define BT_CW2  46   // 2: kc 0..1
#define BT_COUNT 48

static __device__ __forceinline__ v8f zero8() {
  v8f z = {0.f, 0.f, 0.f, 0.f, 0.f, 0.f, 0.f, 0.f};
  return z;
}
static __device__ __forceinline__ int laneid() { return (int)(threadIdx.x & 31u); }

// CDNA5 16-bit A/B operand K-slot mapping (16x32 tile, wave32):
// lanes 0-15: slots 0..7 -> K=0..7, slots 8..15 -> K=16..23
// lanes 16-31: slots 0..7 -> K=8..15, slots 8..15 -> K=24..31
static __device__ __forceinline__ int kslot(int lane, int j) {
  return ((lane & 16) ? 8 : 0) + j + (j & 8);
}

static __device__ __forceinline__ v8f wmma16(v16h a, v16h b, v8f c) {
  return __builtin_amdgcn_wmma_f32_16x16x32_f16(false, a, false, b, (short)0, c,
                                                false, false);
}

// ---- generic (slow-path) B builders: used only by the prep kernel ----------
static __device__ __forceinline__ v16h load_b_w(const float* __restrict__ W,
                                                int ldw, int kbase, int nbase,
                                                int Kmax, int Nmax) {
  int lane = laneid();
  int n = nbase + (lane & 15);
  v16h b;
#pragma unroll
  for (int j = 0; j < 16; ++j) {
    int k = kbase + kslot(lane, j);
    b[j] = (k < Kmax && n < Nmax) ? (_Float16)W[(long long)k * ldw + n]
                                  : (_Float16)0.f;
  }
  return b;
}
static __device__ __forceinline__ v16h load_b_wt(const float* __restrict__ W,
                                                 int ldw, int kbase, int nbase,
                                                 int Kmax, int Nmax) {
  int lane = laneid();
  int n = nbase + (lane & 15);
  v16h b;
#pragma unroll
  for (int j = 0; j < 16; ++j) {
    int k = kbase + kslot(lane, j);
    b[j] = (k < Kmax && n < Nmax) ? (_Float16)W[(long long)n * ldw + k]
                                  : (_Float16)0.f;
  }
  return b;
}

// ---- fast operand loaders --------------------------------------------------
static __device__ __forceinline__ v16h load_bt(const v16h* __restrict__ btab,
                                               int t) {
  return btab[(size_t)t * 32 + laneid()];
}

static __device__ __forceinline__ v16h pack16(float4 f0, float4 f1, float4 f2,
                                              float4 f3) {
  v16h a;
  a[0] = (_Float16)f0.x; a[1] = (_Float16)f0.y; a[2] = (_Float16)f0.z; a[3] = (_Float16)f0.w;
  a[4] = (_Float16)f1.x; a[5] = (_Float16)f1.y; a[6] = (_Float16)f1.z; a[7] = (_Float16)f1.w;
  a[8] = (_Float16)f2.x; a[9] = (_Float16)f2.y; a[10] = (_Float16)f2.z; a[11] = (_Float16)f2.w;
  a[12] = (_Float16)f3.x; a[13] = (_Float16)f3.y; a[14] = (_Float16)f3.z; a[15] = (_Float16)f3.w;
  return a;
}

// A tile, full 32-K chunk from f32 row-major global (lda multiple of 4)
static __device__ __forceinline__ v16h load_a_g32(const float* __restrict__ A,
                                                  int lda, long long row0,
                                                  int kbase) {
  int lane = laneid();
  const float* p = A + (row0 + (lane & 15)) * (long long)lda + kbase +
                   ((lane & 16) ? 8 : 0);
  return pack16(*(const float4*)p, *(const float4*)(p + 4),
                *(const float4*)(p + 16), *(const float4*)(p + 20));
}

// A tile, K=16 matrix (lda == 16), upper half zero
static __device__ __forceinline__ v16h load_a_g16(const float* __restrict__ A,
                                                  long long row0) {
  int lane = laneid();
  const float* p = A + (row0 + (lane & 15)) * 16 + ((lane & 16) ? 8 : 0);
  float4 f0 = *(const float4*)p, f1 = *(const float4*)(p + 4);
  v16h a;
  a[0] = (_Float16)f0.x; a[1] = (_Float16)f0.y; a[2] = (_Float16)f0.z; a[3] = (_Float16)f0.w;
  a[4] = (_Float16)f1.x; a[5] = (_Float16)f1.y; a[6] = (_Float16)f1.z; a[7] = (_Float16)f1.w;
#pragma unroll
  for (int j = 8; j < 16; ++j) a[j] = (_Float16)0.f;
  return a;
}

// A tile from wave-local LDS f16 staging (ld multiple of 8 halves, padded so
// a full 32-K chunk starting at kbase is readable; pad columns hold zeros)
static __device__ __forceinline__ v16h load_a_lds32(const _Float16* S, int ld,
                                                    int kbase) {
  int lane = laneid();
  const _Float16* p = S + (lane & 15) * ld + kbase + ((lane & 16) ? 8 : 0);
  v8h g0 = *(const v8h*)p;
  v8h g1 = *(const v8h*)(p + 16);
  v16h a;
#pragma unroll
  for (int j = 0; j < 8; ++j) { a[j] = g0[j]; a[8 + j] = g1[j]; }
  return a;
}

static __device__ __forceinline__ float sigmoidf(float x) {
  return 1.0f / (1.0f + expf(-x));
}

// ---- kernel 0: pre-swizzle all weight B operands into WMMA lane layout -----
__global__ __launch_bounds__(256) void k_prep_b(
    const float* nw1, const float* nw2, const float* ew1, const float* ew2,
    const float* lin_w, const float* gw1, const float* gw2, const float* wih,
    const float* whh, const float* uw1, const float* uw2, const float* rw1,
    const float* rw2, const float* cw1, const float* cw2,
    v16h* __restrict__ btab) {
  int wv = threadIdx.x >> 5, lane = laneid();
  int t = blockIdx.x * 8 + wv;
  if (t >= BT_COUNT) return;
  v16h b;
  if (t < 10)       b = load_b_w(nw1, NODE_H, (t / 5) * 32, (t % 5) * 16, ATOMD, NODE_H);
  else if (t < 13)  b = load_b_w(nw2, AH, (t - BT_NW2) * 32, 0, NODE_H, AH);
  else if (t < 15)  b = load_b_w(ew1, EDGE_H, 0, (t - BT_EW1) * 16, 16, EDGE_H);
  else if (t < 16)  b = load_b_w(ew2, BH, 0, 0, EDGE_H, BH);
  else if (t < 17)  b = load_b_w(lin_w, AH, 0, 0, BH, AH);
  else if (t < 19)  b = load_b_w(gw1, 32, 0, (t - BT_GW1) * 16, AH, 32);
  else if (t < 20)  b = load_b_w(gw2, AH, 0, 0, 32, AH);
  else if (t < 23)  b = load_b_wt(wih, AH, 0, (t - BT_WIH) * 16, AH, 3 * AH);
  else if (t < 26)  b = load_b_wt(whh, AH, 0, (t - BT_WHH) * 16, AH, 3 * AH);
  else if (t < 28)  b = load_b_w(uw1, 32, 0, (t - BT_UW1) * 16, AH, 32);
  else if (t < 30)  b = load_b_w(uw2, 32, 0, (t - BT_UW2) * 16, 32, 32);
  else if (t < 34)  b = load_b_w(rw1, 64, 0, (t - BT_RW1) * 16, 32, 64);
  else if (t < 42) { int q = t - BT_RW2; b = load_b_w(rw2, 64, (q / 4) * 32, (q % 4) * 16, 64, 64); }
  else if (t < 46)  b = load_b_w(cw1, 64, 0, (t - BT_CW1) * 16, 32, 64);
  else              b = load_b_w(cw2, OUTD, (t - BT_CW2) * 32, 0, 64, OUTD);
  btab[(size_t)t * 32 + lane] = b;
}

// ---- kernel 1: node embed MLP 64->80(relu)->16, LN(16)  -> h ---------------
__global__ __launch_bounds__(256) void k_node_embed(
    const float* __restrict__ x, const v16h* __restrict__ btab,
    const float* nb1, const float* nb2, const float* ng, const float* nbeta,
    float* __restrict__ h) {
  __shared__ __align__(16) _Float16 s_hid[8][16 * 96];  // 80 padded to 96
  __shared__ __align__(16) float    s_ln[8][16 * AH];
  int wv = threadIdx.x >> 5, lane = laneid();
  int tile = blockIdx.x * 8 + wv;
  if (tile >= NN / 16) return;
  int row0 = tile * 16;
  int coln = lane & 15, rb = (lane & 16) ? 8 : 0;
  _Float16* hid = s_hid[wv];
  float* lnr = s_ln[wv];

  v16h a0 = load_a_g32(x, ATOMD, row0, 0);
  v16h a1 = load_a_g32(x, ATOMD, row0, 32);
#pragma unroll
  for (int nt = 0; nt < 5; ++nt) {
    v8f c = wmma16(a0, load_bt(btab, BT_NW1 + nt), zero8());
    c = wmma16(a1, load_bt(btab, BT_NW1 + 5 + nt), c);
    int col = nt * 16 + coln;
    float bias = nb1[col];
#pragma unroll
    for (int i = 0; i < 8; ++i) {
      float v = c[i] + bias;
      hid[(rb + i) * 96 + col] = (_Float16)(v > 0.f ? v : 0.f);
    }
  }
  // zero-fill pad columns 80..95
#pragma unroll
  for (int i = 0; i < 8; ++i) hid[(rb + i) * 96 + 80 + coln] = (_Float16)0.f;

  v8f c2 = wmma16(load_a_lds32(hid, 96, 0), load_bt(btab, BT_NW2 + 0), zero8());
  c2 = wmma16(load_a_lds32(hid, 96, 32), load_bt(btab, BT_NW2 + 1), c2);
  c2 = wmma16(load_a_lds32(hid, 96, 64), load_bt(btab, BT_NW2 + 2), c2);
  {
    float bias = nb2[coln];
#pragma unroll
    for (int i = 0; i < 8; ++i) lnr[(rb + i) * AH + coln] = c2[i] + bias;
  }
  if (lane < 16) {
    const float* rp = lnr + lane * AH;
    float m = 0.f;
    for (int q = 0; q < AH; ++q) m += rp[q];
    m *= (1.f / AH);
    float v = 0.f;
    for (int q = 0; q < AH; ++q) { float d = rp[q] - m; v += d * d; }
    float inv = rsqrtf(v * (1.f / AH) + 1e-5f);
    float* op = h + (long long)(row0 + lane) * AH;
    for (int q = 0; q < AH; ++q) op[q] = (rp[q] - m) * inv * ng[q] + nbeta[q];
  }
}

// ---- kernel 2: edge embed 16->24(relu)->8, LN(8), then 8->16 proj ----------
__global__ __launch_bounds__(256) void k_edge_embed(
    const float* __restrict__ ea, const v16h* __restrict__ btab,
    const float* eb1, const float* eb2, const float* eg, const float* ebeta,
    const float* lin_b, _Float16* __restrict__ eproj) {
  __shared__ __align__(16) _Float16 s_hid[8][16 * 32];  // 24 padded to 32
  __shared__ __align__(16) float    s_lnf[8][16 * BH];
  __shared__ __align__(16) _Float16 s_ean[8][16 * 32];  // 8 padded to 32
  int wv = threadIdx.x >> 5, lane = laneid();
  int tile = blockIdx.x * 8 + wv;
  if (tile >= EE / 16) return;
  int row0 = tile * 16;
  int coln = lane & 15, rb = (lane & 16) ? 8 : 0;
  _Float16* hid = s_hid[wv];
  float* lnf = s_lnf[wv];
  _Float16* ean = s_ean[wv];

  v16h a = load_a_g16(ea, row0);
#pragma unroll
  for (int nt = 0; nt < 2; ++nt) {
    v8f c = wmma16(a, load_bt(btab, BT_EW1 + nt), zero8());
    int col = nt * 16 + coln;
    float bias = (col < EDGE_H) ? eb1[col] : 0.f;
#pragma unroll
    for (int i = 0; i < 8; ++i) {
      float v = c[i] + bias;
      v = v > 0.f ? v : 0.f;
      hid[(rb + i) * 32 + col] = (_Float16)((col < EDGE_H) ? v : 0.f);
    }
  }
  v8f c2 = wmma16(load_a_lds32(hid, 32, 0), load_bt(btab, BT_EW2), zero8());
  if (coln < BH) {
    float bias = eb2[coln];
#pragma unroll
    for (int i = 0; i < 8; ++i) lnf[(rb + i) * BH + coln] = c2[i] + bias;
  }
  if (lane < 16) {
    const float* rp = lnf + lane * BH;
    float m = 0.f;
    for (int q = 0; q < BH; ++q) m += rp[q];
    m *= (1.f / BH);
    float v = 0.f;
    for (int q = 0; q < BH; ++q) { float d = rp[q] - m; v += d * d; }
    float inv = rsqrtf(v * (1.f / BH) + 1e-5f);
    for (int q = 0; q < BH; ++q)
      ean[lane * 32 + q] = (_Float16)((rp[q] - m) * inv * eg[q] + ebeta[q]);
    for (int q = BH; q < 32; ++q) ean[lane * 32 + q] = (_Float16)0.f;
  }
  v8f c3 = wmma16(load_a_lds32(ean, 32, 0), load_bt(btab, BT_LIN), zero8());
  {
    float bias = lin_b[coln];
#pragma unroll
    for (int i = 0; i < 8; ++i)
      eproj[(long long)(row0 + rb + i) * AH + coln] = (_Float16)(c3[i] + bias);
  }
}

// ---- kernel 3: edge message + scatter: agg[dst] += relu(h[src]+eproj) ------
__global__ __launch_bounds__(256) void k_message(const int* __restrict__ ei,
                                                 const float* __restrict__ h,
                                                 const _Float16* __restrict__ eproj,
                                                 float* __restrict__ agg) {
  long long e = (long long)blockIdx.x * blockDim.x + threadIdx.x;
  if (e >= EE) return;
  int s = ei[e], d = ei[EE + e];
  const float4* hs = (const float4*)(h + (long long)s * AH);
  float4 h0 = hs[0], h1 = hs[1], h2 = hs[2], h3 = hs[3];
  v8h e0 = *(const v8h*)(eproj + e * AH);
  v8h e1 = *(const v8h*)(eproj + e * AH + 8);
  float msg[16];
  msg[0] = h0.x + (float)e0[0]; msg[1] = h0.y + (float)e0[1];
  msg[2] = h0.z + (float)e0[2]; msg[3] = h0.w + (float)e0[3];
  msg[4] = h1.x + (float)e0[4]; msg[5] = h1.y + (float)e0[5];
  msg[6] = h1.z + (float)e0[6]; msg[7] = h1.w + (float)e0[7];
  msg[8] = h2.x + (float)e1[0]; msg[9] = h2.y + (float)e1[1];
  msg[10] = h2.z + (float)e1[2]; msg[11] = h2.w + (float)e1[3];
  msg[12] = h3.x + (float)e1[4]; msg[13] = h3.y + (float)e1[5];
  msg[14] = h3.z + (float)e1[6]; msg[15] = h3.w + (float)e1[7];
  float* ad = agg + (long long)d * AH;
#pragma unroll
  for (int q = 0; q < AH; ++q) {
    float m = msg[q] > 0.f ? msg[q] : 0.f;
    atomicAdd(ad + q, m);
  }
}

// ---- kernel 4: GINE inner MLP 16->32(relu)->16 + global-LN partial sums ----
__global__ __launch_bounds__(256) void k_gine(
    const float* __restrict__ h, const float* __restrict__ agg,
    const v16h* __restrict__ btab, const float* gb1, const float* gb2,
    float* __restrict__ z, float* __restrict__ red2) {
  __shared__ __align__(16) _Float16 s_hid[8][16 * 32];
  int wv = threadIdx.x >> 5, lane = laneid();
  int tile = blockIdx.x * 8 + wv;
  if (tile >= NN / 16) return;
  int row0 = tile * 16;
  int coln = lane & 15, rb = (lane & 16) ? 8 : 0;
  _Float16* hid = s_hid[wv];

  // A = (h + agg), K = 16 zero-padded to 32
  v16h a;
  {
    long long roff = (long long)(row0 + coln) * AH + ((lane & 16) ? 8 : 0);
    const float4* ph = (const float4*)(h + roff);
    const float4* pa = (const float4*)(agg + roff);
    float4 h0 = ph[0], h1 = ph[1], q0 = pa[0], q1 = pa[1];
    a[0] = (_Float16)(h0.x + q0.x); a[1] = (_Float16)(h0.y + q0.y);
    a[2] = (_Float16)(h0.z + q0.z); a[3] = (_Float16)(h0.w + q0.w);
    a[4] = (_Float16)(h1.x + q1.x); a[5] = (_Float16)(h1.y + q1.y);
    a[6] = (_Float16)(h1.z + q1.z); a[7] = (_Float16)(h1.w + q1.w);
#pragma unroll
    for (int j = 8; j < 16; ++j) a[j] = (_Float16)0.f;
  }
#pragma unroll
  for (int nt = 0; nt < 2; ++nt) {
    v8f c = wmma16(a, load_bt(btab, BT_GW1 + nt), zero8());
    int col = nt * 16 + coln;
    float bias = gb1[col];
#pragma unroll
    for (int i = 0; i < 8; ++i) {
      float v = c[i] + bias;
      hid[(rb + i) * 32 + col] = (_Float16)(v > 0.f ? v : 0.f);
    }
  }
  v8f c2 = wmma16(load_a_lds32(hid, 32, 0), load_bt(btab, BT_GW2), zero8());
  float bias = gb2[coln];
  float ps = 0.f, ps2 = 0.f;
#pragma unroll
  for (int i = 0; i < 8; ++i) {
    float v = c2[i] + bias;
    z[(long long)(row0 + rb + i) * AH + coln] = v;
    ps += v;
    ps2 += v * v;
  }
#pragma unroll
  for (int off = 16; off > 0; off >>= 1) {
    ps += __shfl_down(ps, off);
    ps2 += __shfl_down(ps2, off);
  }
  if (lane == 0) {
    atomicAdd(&red2[0], ps);
    atomicAdd(&red2[1], ps2);
  }
}

// ---- kernel 5: global graph-LN + GRU cell (updates h in place) -------------
__global__ __launch_bounds__(256) void k_gru(
    const float* __restrict__ z, float* __restrict__ h,
    const v16h* __restrict__ btab, const float* bih, const float* bhh,
    const float* ngm, const float* nbt, const float* __restrict__ red2) {
  int wv = threadIdx.x >> 5, lane = laneid();
  int tile = blockIdx.x * 8 + wv;
  if (tile >= NN / 16) return;
  int row0 = tile * 16;
  int coln = lane & 15, rb = (lane & 16) ? 8 : 0;
  int off = (lane & 16) ? 8 : 0;

  const float invM = 1.f / (16.f * (float)NN);
  float m = red2[0] * invM;
  float var = red2[1] * invM - m * m;
  float inv = rsqrtf(var + 1e-5f);

  // A = graph-LN(z) with per-channel affine, K=16 padded to 32
  v16h az;
  {
    const float* pz = z + (long long)(row0 + coln) * AH + off;
    float4 z0 = *(const float4*)pz, z1 = *(const float4*)(pz + 4);
    float4 g0 = *(const float4*)(ngm + off), g1 = *(const float4*)(ngm + off + 4);
    float4 b0 = *(const float4*)(nbt + off), b1 = *(const float4*)(nbt + off + 4);
    az[0] = (_Float16)((z0.x - m) * inv * g0.x + b0.x);
    az[1] = (_Float16)((z0.y - m) * inv * g0.y + b0.y);
    az[2] = (_Float16)((z0.z - m) * inv * g0.z + b0.z);
    az[3] = (_Float16)((z0.w - m) * inv * g0.w + b0.w);
    az[4] = (_Float16)((z1.x - m) * inv * g1.x + b1.x);
    az[5] = (_Float16)((z1.y - m) * inv * g1.y + b1.y);
    az[6] = (_Float16)((z1.z - m) * inv * g1.z + b1.z);
    az[7] = (_Float16)((z1.w - m) * inv * g1.w + b1.w);
#pragma unroll
    for (int j = 8; j < 16; ++j) az[j] = (_Float16)0.f;
  }
  v16h ahh = load_a_g16(h, row0);

  v8f gir = wmma16(az, load_bt(btab, BT_WIH + 0), zero8());
  v8f giz = wmma16(az, load_bt(btab, BT_WIH + 1), zero8());
  v8f gin = wmma16(az, load_bt(btab, BT_WIH + 2), zero8());
  v8f ghr = wmma16(ahh, load_bt(btab, BT_WHH + 0), zero8());
  v8f ghz = wmma16(ahh, load_bt(btab, BT_WHH + 1), zero8());
  v8f ghn = wmma16(ahh, load_bt(btab, BT_WHH + 2), zero8());

  float bir = bih[coln], biz = bih[16 + coln], bin = bih[32 + coln];
  float bhr = bhh[coln], bhz = bhh[16 + coln], bhn = bhh[32 + coln];
#pragma unroll
  for (int i = 0; i < 8; ++i) {
    long long idx = (long long)(row0 + rb + i) * AH + coln;
    float hv = h[idx];
    float rg = sigmoidf(gir[i] + bir + ghr[i] + bhr);
    float zg = sigmoidf(giz[i] + biz + ghz[i] + bhz);
    float nn = tanhf(gin[i] + bin + rg * (ghn[i] + bhn));
    h[idx] = (1.f - zg) * nn + zg * hv;
  }
}

// ---- kernel 6: updim 16->32(leaky)->32, LN(32) -> u; pooled accumulation ---
__global__ __launch_bounds__(256) void k_updim(
    const float* __restrict__ h, const int* __restrict__ batch,
    const v16h* __restrict__ btab, const float* ub1, const float* ub2,
    const float* ug, const float* ubeta, float* __restrict__ u,
    float* __restrict__ pooled, float* __restrict__ cnt) {
  __shared__ __align__(16) _Float16 s_hid[8][16 * 32];
  __shared__ __align__(16) float    s_ln[8][16 * 32];
  int wv = threadIdx.x >> 5, lane = laneid();
  int tile = blockIdx.x * 8 + wv;
  if (tile >= NN / 16) return;
  int row0 = tile * 16;
  int coln = lane & 15, rb = (lane & 16) ? 8 : 0;
  _Float16* hid = s_hid[wv];
  float* lnr = s_ln[wv];

  v16h a = load_a_g16(h, row0);
#pragma unroll
  for (int nt = 0; nt < 2; ++nt) {
    v8f c = wmma16(a, load_bt(btab, BT_UW1 + nt), zero8());
    int col = nt * 16 + coln;
    float bias = ub1[col];
#pragma unroll
    for (int i = 0; i < 8; ++i) {
      float v = c[i] + bias;
      hid[(rb + i) * 32 + col] = (_Float16)(v > 0.f ? v : 0.01f * v);
    }
  }
  v16h a2 = load_a_lds32(hid, 32, 0);
#pragma unroll
  for (int nt = 0; nt < 2; ++nt) {
    v8f c2 = wmma16(a2, load_bt(btab, BT_UW2 + nt), zero8());
    int col = nt * 16 + coln;
    float bias = ub2[col];
#pragma unroll
    for (int i = 0; i < 8; ++i) lnr[(rb + i) * 32 + col] = c2[i] + bias;
  }
  if (lane < 16) {
    const float* rp = lnr + lane * 32;
    float mm = 0.f;
    for (int q = 0; q < 32; ++q) mm += rp[q];
    mm *= (1.f / 32.f);
    float v = 0.f;
    for (int q = 0; q < 32; ++q) { float d = rp[q] - mm; v += d * d; }
    float inv = rsqrtf(v * (1.f / 32.f) + 1e-5f);
    int row = row0 + lane;
    int g = batch[row];
    float* op = u + (long long)row * 32;
    for (int q = 0; q < 32; ++q) {
      float val = (rp[q] - mm) * inv * ug[q] + ubeta[q];
      op[q] = val;
      atomicAdd(&pooled[(long long)g * 32 + q], val);
    }
    atomicAdd(&cnt[g], 1.f);
  }
}

// ---- kernel 7: reconstruct head 32->64(relu)->64 ---------------------------
__global__ __launch_bounds__(256) void k_rec(const float* __restrict__ u,
                                             const v16h* __restrict__ btab,
                                             const float* rb1, const float* rb2,
                                             float* __restrict__ rec) {
  __shared__ __align__(16) _Float16 s_hid[8][16 * 64];
  int wv = threadIdx.x >> 5, lane = laneid();
  int tile = blockIdx.x * 8 + wv;
  if (tile >= NN / 16) return;
  int row0 = tile * 16;
  int coln = lane & 15, rb = (lane & 16) ? 8 : 0;
  _Float16* hid = s_hid[wv];

  v16h a = load_a_g32(u, 32, row0, 0);
#pragma unroll
  for (int nt = 0; nt < 4; ++nt) {
    v8f c = wmma16(a, load_bt(btab, BT_RW1 + nt), zero8());
    int col = nt * 16 + coln;
    float bias = rb1[col];
#pragma unroll
    for (int i = 0; i < 8; ++i) {
      float v = c[i] + bias;
      hid[(rb + i) * 64 + col] = (_Float16)(v > 0.f ? v : 0.f);
    }
  }
  v16h a0 = load_a_lds32(hid, 64, 0);
  v16h a1 = load_a_lds32(hid, 64, 32);
#pragma unroll
  for (int nt = 0; nt < 4; ++nt) {
    v8f c2 = wmma16(a0, load_bt(btab, BT_RW2 + nt), zero8());
    c2 = wmma16(a1, load_bt(btab, BT_RW2 + 4 + nt), c2);
    int col = nt * 16 + coln;
    float bias = rb2[col];
#pragma unroll
    for (int i = 0; i < 8; ++i)
      rec[(long long)(row0 + rb + i) * 64 + col] = c2[i] + bias;
  }
}

// ---- kernel 8: per-graph mean + classifier 32->64(relu)->2 -----------------
__global__ __launch_bounds__(256) void k_clf(const float* __restrict__ pooled,
                                             const float* __restrict__ cnt,
                                             const v16h* __restrict__ btab,
                                             const float* cb1,
                                             float* __restrict__ out) {
  __shared__ __align__(16) _Float16 s_hid[8][16 * 64];
  int wv = threadIdx.x >> 5, lane = laneid();
  int tile = blockIdx.x * 8 + wv;
  int ntiles = (GG + 15) / 16;
  if (tile >= ntiles) return;
  int row0 = tile * 16;
  int coln = lane & 15, rb = (lane & 16) ? 8 : 0;
  _Float16* hid = s_hid[wv];

  v16h a;
  {
    int r = row0 + coln;
    int rc = r < GG ? r : GG - 1;
    float cv = cnt[rc];
    cv = cv > 1.f ? cv : 1.f;
    float invc = 1.f / cv;
    const float* pp = pooled + (long long)rc * 32 + ((lane & 16) ? 8 : 0);
    float4 p0 = *(const float4*)pp, p1 = *(const float4*)(pp + 4);
    float4 p2 = *(const float4*)(pp + 16), p3 = *(const float4*)(pp + 20);
    p0.x *= invc; p0.y *= invc; p0.z *= invc; p0.w *= invc;
    p1.x *= invc; p1.y *= invc; p1.z *= invc; p1.w *= invc;
    p2.x *= invc; p2.y *= invc; p2.z *= invc; p2.w *= invc;
    p3.x *= invc; p3.y *= invc; p3.z *= invc; p3.w *= invc;
    a = pack16(p0, p1, p2, p3);
  }
#pragma unroll
  for (int nt = 0; nt < 4; ++nt) {
    v8f c = wmma16(a, load_bt(btab, BT_CW1 + nt), zero8());
    int col = nt * 16 + coln;
    float bias = cb1[col];
#pragma unroll
    for (int i = 0; i < 8; ++i) {
      float v = c[i] + bias;
      hid[(rb + i) * 64 + col] = (_Float16)(v > 0.f ? v : 0.f);
    }
  }
  v8f c2 = wmma16(load_a_lds32(hid, 64, 0), load_bt(btab, BT_CW2 + 0), zero8());
  c2 = wmma16(load_a_lds32(hid, 64, 32), load_bt(btab, BT_CW2 + 1), c2);
  if (coln < OUTD) {
#pragma unroll
    for (int i = 0; i < 8; ++i) {
      int g = row0 + rb + i;
      if (g < GG) out[(long long)g * OUTD + coln] = c2[i];
    }
  }
}

// ---- launch ---------------------------------------------------------------
extern "C" void kernel_launch(void* const* d_in, const int* in_sizes, int n_in,
                              void* d_out, int out_size, void* d_ws,
                              size_t ws_size, hipStream_t stream) {
  const float* x     = (const float*)d_in[0];
  const int*   ei    = (const int*)d_in[1];
  const float* ea    = (const float*)d_in[2];
  const int*   batch = (const int*)d_in[3];
  const float *nw1 = (const float*)d_in[4], *nb1 = (const float*)d_in[5],
              *nw2 = (const float*)d_in[6], *nb2 = (const float*)d_in[7],
              *ng = (const float*)d_in[8], *nbeta = (const float*)d_in[9];
  const float *ew1 = (const float*)d_in[10], *eb1 = (const float*)d_in[11],
              *ew2 = (const float*)d_in[12], *eb2 = (const float*)d_in[13],
              *eg = (const float*)d_in[14], *ebeta = (const float*)d_in[15];
  const float *lin_w = (const float*)d_in[16], *lin_b = (const float*)d_in[17];
  const float *gw1 = (const float*)d_in[18], *gb1 = (const float*)d_in[19],
              *gw2 = (const float*)d_in[20], *gb2 = (const float*)d_in[21];
  const float *normg = (const float*)d_in[22], *normb = (const float*)d_in[23];
  const float *wih = (const float*)d_in[24], *whh = (const float*)d_in[25],
              *bih = (const float*)d_in[26], *bhh = (const float*)d_in[27];
  const float *uw1 = (const float*)d_in[28], *ub1 = (const float*)d_in[29],
              *uw2 = (const float*)d_in[30], *ub2 = (const float*)d_in[31],
              *ug = (const float*)d_in[32], *ubeta = (const float*)d_in[33];
  const float *cw1 = (const float*)d_in[34], *cb1 = (const float*)d_in[35],
              *cw2 = (const float*)d_in[36];
  const float *rw1 = (const float*)d_in[37], *rb1 = (const float*)d_in[38],
              *rw2 = (const float*)d_in[39], *rb2 = (const float*)d_in[40];

  char* wsp = (char*)d_ws;
  float* h = (float*)wsp;        wsp += (size_t)NN * AH * sizeof(float);
  float* agg = (float*)wsp;      wsp += (size_t)NN * AH * sizeof(float);
  float* zb = (float*)wsp;       wsp += (size_t)NN * AH * sizeof(float);
  float* ubuf = (float*)wsp;     wsp += (size_t)NN * 2 * AH * sizeof(float);
  float* pooled = (float*)wsp;   wsp += (size_t)GG * 2 * AH * sizeof(float);
  float* cnt = (float*)wsp;      wsp += (size_t)GG * sizeof(float);
  float* red = (float*)wsp;      wsp += 8 * sizeof(float);
  v16h* btab = (v16h*)wsp;       wsp += (size_t)BT_COUNT * 32 * sizeof(v16h);
  _Float16* eproj = (_Float16*)wsp;  // E*AH f16 (128 MB)

  float* out = (float*)d_out;
  float* rec = out + (size_t)GG * OUTD;

  hipMemsetAsync(red, 0, 8 * sizeof(float), stream);
  hipMemsetAsync(pooled, 0, (size_t)GG * 2 * AH * sizeof(float), stream);
  hipMemsetAsync(cnt, 0, (size_t)GG * sizeof(float), stream);

  dim3 blk(256);
  int nt_nodes = NN / 16;   // 31250
  int nt_edges = EE / 16;   // 250000
  int nb_nodes = (nt_nodes + 7) / 8;
  int nb_edges = (nt_edges + 7) / 8;

  k_prep_b<<<(BT_COUNT + 7) / 8, blk, 0, stream>>>(nw1, nw2, ew1, ew2, lin_w,
                                                   gw1, gw2, wih, whh, uw1, uw2,
                                                   rw1, rw2, cw1, cw2, btab);
  k_node_embed<<<nb_nodes, blk, 0, stream>>>(x, btab, nb1, nb2, ng, nbeta, h);
  k_edge_embed<<<nb_edges, blk, 0, stream>>>(ea, btab, eb1, eb2, eg, ebeta,
                                             lin_b, eproj);
  for (int l = 0; l < NLAYER; ++l) {
    hipMemsetAsync(agg, 0, (size_t)NN * AH * sizeof(float), stream);
    k_message<<<(EE + 255) / 256, blk, 0, stream>>>(ei, h, eproj, agg);
    k_gine<<<nb_nodes, blk, 0, stream>>>(h, agg, btab, gb1, gb2, zb,
                                         red + 2 * l);
    k_gru<<<nb_nodes, blk, 0, stream>>>(zb, h, btab, bih, bhh, normg + l * AH,
                                        normb + l * AH, red + 2 * l);
  }
  k_updim<<<nb_nodes, blk, 0, stream>>>(h, batch, btab, ub1, ub2, ug, ubeta,
                                        ubuf, pooled, cnt);
  k_rec<<<nb_nodes, blk, 0, stream>>>(ubuf, btab, rb1, rb2, rec);
  k_clf<<<(((GG + 15) / 16) + 7) / 8, blk, 0, stream>>>(pooled, cnt, btab, cb1,
                                                        out);
}
